// SuperGlobalRerank_49426483642969
// MI455X (gfx1250) — compile-verified, matrix operand order checked by reading, not run on hardware
//
#include <hip/hip_runtime.h>

// ---------------- problem constants (from reference) ----------------
#define B_Q      128
#define D_DIM    512
#define N_DB     100000
#define M_TOP    10
#define TKK      10          // top-k kept everywhere (K == M == 10)
#define NCHUNKS  4
#define CHUNK_N  25000       // 4 * 25000 == 100000
#define TOP_X    3
#define BETA_W   2.0f
#define EPS_W    1e-12f

#define HAVE_TDM __has_builtin(__builtin_amdgcn_tensor_load_to_lds)

typedef __attribute__((ext_vector_type(16))) __bf16          v16bf;
typedef __attribute__((ext_vector_type(16))) unsigned short  v16u;
typedef __attribute__((ext_vector_type(8)))  unsigned short  u16x8;
typedef __attribute__((ext_vector_type(4)))  unsigned short  u16x4;
typedef __attribute__((ext_vector_type(8)))  float           v8f;
typedef __attribute__((ext_vector_type(4)))  float           f32x4;
typedef __attribute__((ext_vector_type(4)))  unsigned int    u32x4;
typedef __attribute__((ext_vector_type(8)))  int             i32x8;
typedef __attribute__((ext_vector_type(4)))  int             i32x4;

typedef __attribute__((address_space(3))) unsigned short lds_u16;

__device__ __forceinline__ unsigned short f2bf_rne(float f) {
  unsigned int x = __float_as_uint(f);
  unsigned int r = x + 0x7FFFu + ((x >> 16) & 1u);
  return (unsigned short)(r >> 16);
}

__device__ __forceinline__ unsigned lds_byte_addr(unsigned short* p) {
  return (unsigned)(uintptr_t)(lds_u16*)p;   // generic -> LDS addrspace offset
}

#if HAVE_TDM
// 2-D Tensor-DMA descriptor load: tile (tile_d1 rows x tile_d0 elems) of a
// (tensor_d1 x tensor_d0) bf16 tensor -> LDS, with 16B pad after every 64B row
// (LDS row stride = 40 elements = 80 B).  See CDNA5 ISA ch.10.7 / D# spec.
__device__ __forceinline__ void tdm_load_2d(unsigned lds_addr, const unsigned short* gptr,
                                            unsigned tensor_d0, unsigned tensor_d1,
                                            unsigned tile_d0, unsigned tile_d1,
                                            unsigned stride0_elems) {
  unsigned long long ga = (unsigned long long)(uintptr_t)gptr;
  u32x4 g0;
  g0[0] = 1u;                                      // count=1, user descriptor
  g0[1] = lds_addr;                                // LDS byte address
  g0[2] = (unsigned)(ga & 0xFFFFFFFFu);            // global_addr[31:0]
  g0[3] = (unsigned)((ga >> 32) & 0x01FFFFFFu)     // global_addr[56:32]
        | (2u << 30);                              // type = 2 ("image")
  i32x8 g1;
  g1[0] = (int)((1u << 16)                         // data_size = 2 bytes
              | (1u << 20)                         // pad_enable
              | (3u << 22)                         // pad_interval = 16 dwords (64B)
              | (3u << 25));                       // pad_amount   = 4 dwords (16B)
  g1[1] = (int)((tensor_d0 & 0xFFFFu) << 16);      // tensor_dim0[15:0]
  g1[2] = (int)((tensor_d0 >> 16) | ((tensor_d1 & 0xFFFFu) << 16));
  g1[3] = (int)((tensor_d1 >> 16) | (tile_d0 << 16));
  g1[4] = (int)(tile_d1 & 0xFFFFu);                // tile_dim1 (tile_dim2 = 0)
  g1[5] = (int)stride0_elems;                      // tensor_dim0_stride[31:0]
  g1[6] = 0;
  g1[7] = 0;
  i32x4 z4 = {0, 0, 0, 0};
#if __clang_major__ >= 23
  i32x8 z8 = {0, 0, 0, 0, 0, 0, 0, 0};
  __builtin_amdgcn_tensor_load_to_lds(g0, g1, z4, z4, z8, 0);
#else
  __builtin_amdgcn_tensor_load_to_lds(g0, g1, z4, z4, 0);
#endif
}
#endif

// ---------------- 1. f32 -> bf16 conversion ----------------
__global__ void f32_to_bf16_kernel(const float* __restrict__ src,
                                   unsigned short* __restrict__ dst, long n) {
  long i = ((long)blockIdx.x * blockDim.x + threadIdx.x) * 4;
  if (i + 3 < n) {
    f32x4 v = *(const f32x4*)(src + i);
    u16x4 o;
    o[0] = f2bf_rne(v[0]); o[1] = f2bf_rne(v[1]);
    o[2] = f2bf_rne(v[2]); o[3] = f2bf_rne(v[3]);
    *(u16x4*)(dst + i) = o;
  } else {
    for (long j = i; j < n; ++j) dst[j] = f2bf_rne(src[j]);
  }
}

// ---------------- 2. WMMA bf16 GEMM with TDM double-buffered LDS pipeline ----
// S = A(M x 512) * DB(N x 512)^T.  Block: 256 thr = 8 waves, tile 32M x 256N.
// Wave tile 16x64 via 4 x v_wmma_f32_16x16x32_bf16.  Per 32-wide K step the
// TDM (wave 0) DMAs the next 256x32 B tile + 32x32 A tile into the spare LDS
// buffer (padded to 80B row stride) while all waves run WMMAs on the current
// one; sync = s_wait_tensorcnt(0) + workgroup barrier.
__global__ void wmma_gemm_nt_bf16(const unsigned short* __restrict__ A, int Mrows,
                                  const unsigned short* __restrict__ Bdb, int Ncols,
                                  float* __restrict__ S) {
  __shared__ __align__(16) unsigned short shB[2][256 * 40];  // 40 KB
  __shared__ __align__(16) unsigned short shA[2][32 * 40];   //  5 KB
  const int K = D_DIM;
  int tid  = threadIdx.x;
  int wave = tid >> 5;
  int lane = tid & 31;
  int wm = wave >> 2;                 // 0..1
  int wn = wave & 3;                  // 0..3
  int mBase = blockIdx.y * 32;
  int nBase = blockIdx.x * 256;
  if (mBase >= Mrows) return;
  int h  = lane >> 4;
  int ml = lane & 15;

  auto stage = [&](int buf, int k0) {
#if HAVE_TDM
    if (wave == 0) {
      tdm_load_2d(lds_byte_addr(&shB[buf][0]), Bdb + (size_t)nBase * K + k0,
                  K, (unsigned)Ncols, 32, 256, K);
      tdm_load_2d(lds_byte_addr(&shA[buf][0]), A + (size_t)mBase * K + k0,
                  K, (unsigned)Mrows, 32, 32, K);
    }
#else
    for (int idx = tid; idx < 256 * 4; idx += 256) {     // B tile: 256 rows x 64B
      int r = idx >> 2, sg = idx & 3;
      int n = nBase + r;
      u16x8 v = {};
      if (n < Ncols) v = *(const u16x8*)(Bdb + (size_t)n * K + k0 + sg * 8);
      *(u16x8*)&shB[buf][r * 40 + sg * 8] = v;
    }
    if (tid < 32 * 4) {                                  // A tile: 32 rows x 64B
      int r = tid >> 2, sg = tid & 3;
      *(u16x8*)&shA[buf][r * 40 + sg * 8] =
          *(const u16x8*)(A + (size_t)(mBase + r) * K + k0 + sg * 8);
    }
#endif
  };
  auto wait_stage = [&]() {
#if HAVE_TDM
    if (wave == 0) __builtin_amdgcn_s_wait_tensorcnt(0);
#endif
    __syncthreads();
  };

  v8f acc[4] = {};
  stage(0, 0);
  wait_stage();
  int buf = 0;
  for (int k0 = 0; k0 < K; k0 += 32) {
    if (k0 + 32 < K) stage(buf ^ 1, k0 + 32);   // overlap DMA with WMMA

    int arow = wm * 16 + ml;
    u16x8 alo = *(const u16x8*)&shA[buf][arow * 40 + 8 * h];
    u16x8 ahi = *(const u16x8*)&shA[buf][arow * 40 + 16 + 8 * h];
    v16u  av  = __builtin_shufflevector(alo, ahi, 0,1,2,3,4,5,6,7,8,9,10,11,12,13,14,15);
    v16bf afrag = __builtin_bit_cast(v16bf, av);
#pragma unroll
    for (int j = 0; j < 4; ++j) {
      int brow = wn * 64 + j * 16 + ml;
      u16x8 blo = *(const u16x8*)&shB[buf][brow * 40 + 8 * h];
      u16x8 bhi = *(const u16x8*)&shB[buf][brow * 40 + 16 + 8 * h];
      v16u  bv  = __builtin_shufflevector(blo, bhi, 0,1,2,3,4,5,6,7,8,9,10,11,12,13,14,15);
      v16bf bfrag = __builtin_bit_cast(v16bf, bv);
      acc[j] = __builtin_amdgcn_wmma_f32_16x16x32_bf16(
          false, afrag, false, bfrag, (short)0, acc[j], false, false);
    }

    wait_stage();                               // next buffer ready, all reads done
    buf ^= 1;
  }

  // C/D layout: lane l (n = l&15), VGPR r -> M = r + 8*(l>>4)
#pragma unroll
  for (int j = 0; j < 4; ++j) {
    int n = nBase + wn * 64 + j * 16 + ml;
    if (n < Ncols) {
#pragma unroll
      for (int r = 0; r < 8; ++r) {
        int m = mBase + wm * 16 + r + 8 * h;
        S[(long)m * Ncols + n] = acc[j][r];
      }
    }
  }
}

// ---------------- 3. per-chunk top-10 (one block per row) ----------------
__global__ void chunk_topk_kernel(const float* __restrict__ S, int ncols, int colOffset,
                                  int chunkIdx, int nchunks,
                                  float* __restrict__ candS, int* __restrict__ candI) {
  __shared__ float shS[256 * TKK];
  __shared__ int   shI[256 * TKK];
  int row = blockIdx.x;
  int tid = threadIdx.x;
  const float* s = S + (long)row * ncols;

  float bs[TKK]; int bi[TKK];
#pragma unroll
  for (int k = 0; k < TKK; ++k) { bs[k] = -3.0e38f; bi[k] = -1; }

  for (int c = tid; c < ncols; c += 256) {
    float v  = s[c];
    int   id = colOffset + c;
    if (v > bs[TKK - 1]) {
#pragma unroll
      for (int k = 0; k < TKK; ++k) {       // static-index bubble insert
        if (v > bs[k]) { float tv = bs[k]; int ti = bi[k];
                         bs[k] = v; bi[k] = id; v = tv; id = ti; }
      }
    }
  }

  for (int stride = 128; stride >= 1; stride >>= 1) {
    if (tid < 2 * stride) {
#pragma unroll
      for (int k = 0; k < TKK; ++k) { shS[tid * TKK + k] = bs[k]; shI[tid * TKK + k] = bi[k]; }
    }
    __syncthreads();
    if (tid < stride) {
#pragma unroll
      for (int k = 0; k < TKK; ++k) {
        float v  = shS[(tid + stride) * TKK + k];
        int   id = shI[(tid + stride) * TKK + k];
#pragma unroll
        for (int t = 0; t < TKK; ++t) {
          if (v > bs[t]) { float tv = bs[t]; int ti = bi[t];
                           bs[t] = v; bi[t] = id; v = tv; id = ti; }
        }
      }
    }
    __syncthreads();
  }

  if (tid == 0) {
    long base = ((long)row * nchunks + chunkIdx) * TKK;
#pragma unroll
    for (int k = 0; k < TKK; ++k) { candS[base + k] = bs[k]; candI[base + k] = bi[k]; }
  }
}

// ---------------- 4. merge per-chunk candidates -> global top-10 ----------------
__global__ void merge_cand_kernel(const float* __restrict__ candS, const int* __restrict__ candI,
                                  int nrows, int nch,
                                  float* __restrict__ outS, int* __restrict__ outI) {
  int row = blockIdx.x * blockDim.x + threadIdx.x;
  if (row >= nrows) return;
  float bs[TKK]; int bi[TKK];
#pragma unroll
  for (int k = 0; k < TKK; ++k) { bs[k] = -3.0e38f; bi[k] = -1; }
  long base = (long)row * nch * TKK;
  for (int j = 0; j < nch * TKK; ++j) {
    float v = candS[base + j]; int id = candI[base + j];
#pragma unroll
    for (int t = 0; t < TKK; ++t) {
      if (v > bs[t]) { float tv = bs[t]; int ti = bi[t];
                       bs[t] = v; bi[t] = id; v = tv; id = ti; }
    }
  }
#pragma unroll
  for (int k = 0; k < TKK; ++k) {
    outS[(long)row * TKK + k] = bs[k];
    outI[(long)row * TKK + k] = bi[k];
  }
}

// ---------------- 5. query_top_k = L2-normalized max(q, top9 db rows) ----------------
__global__ void build_qtk_kernel(const float* __restrict__ q, const float* __restrict__ db,
                                 const int* __restrict__ topmI, float* __restrict__ qtk) {
  __shared__ float red[256];
  __shared__ float s_norm;
  int b = blockIdx.x, tid = threadIdx.x;
  int ids[TKK - 1];
#pragma unroll
  for (int m = 0; m < TKK - 1; ++m) ids[m] = topmI[b * TKK + m];
  float vals[2]; float ssq = 0.0f;
#pragma unroll
  for (int t = 0; t < 2; ++t) {
    int d = tid + t * 256;
    float v = q[(long)b * D_DIM + d];
#pragma unroll
    for (int m = 0; m < TKK - 1; ++m) v = fmaxf(v, db[(long)ids[m] * D_DIM + d]);
    vals[t] = v; ssq += v * v;
  }
  red[tid] = ssq; __syncthreads();
  for (int s = 128; s > 0; s >>= 1) { if (tid < s) red[tid] += red[tid + s]; __syncthreads(); }
  if (tid == 0) s_norm = fmaxf(sqrtf(red[0]), EPS_W);
  __syncthreads();
#pragma unroll
  for (int t = 0; t < 2; ++t)
    qtk[(long)b * D_DIM + tid + t * 256] = vals[t] / s_norm;
}

// ---------------- 6. gather top_m rows as bf16 A-matrix for GEMM2 ----------------
__global__ void build_a2_kernel(const float* __restrict__ db, const int* __restrict__ topmI,
                                unsigned short* __restrict__ A2) {
  int row = blockIdx.x;              // 0..1279
  int id  = topmI[row];
  for (int d = threadIdx.x; d < D_DIM; d += 256)
    A2[(long)row * D_DIM + d] = f2bf_rne(db[(long)id * D_DIM + d]);
}

// ---------------- 7. refine + final per-(b,m) score ----------------
__global__ void refine_score_kernel(const float* __restrict__ q, const float* __restrict__ qtk,
                                    const float* __restrict__ db,
                                    const float* __restrict__ s2, const int* __restrict__ i2,
                                    float* __restrict__ finalS) {
  __shared__ float rA[256], rB[256], rC[256];
  int row = blockIdx.x;              // b*10 + m
  int b   = row / M_TOP;
  int tid = threadIdx.x;

  float w[TKK]; int ids[TKK]; float sumw = 0.0f;
#pragma unroll
  for (int k = 0; k < TKK; ++k) {
    float sc = s2[(long)row * TKK + k];
    w[k] = sc * BETA_W; sumw += w[k];
    ids[k] = i2[(long)row * TKK + k];
  }
  float normf = 1.0f + BETA_W + sumw;   // 1 + sum(weights), weights=[1,s...]*BETA

  float ssq = 0.0f, dq = 0.0f, dqt = 0.0f;
#pragma unroll
  for (int t = 0; t < 2; ++t) {
    int d = tid + t * 256;
    float qv = q[(long)b * D_DIM + d];
    float acc = BETA_W * qv;
#pragma unroll
    for (int k = 0; k < TKK; ++k) acc += w[k] * db[(long)ids[k] * D_DIM + d];
    float r = acc / normf;
    ssq += r * r;
    dq  += r * qv;
    dqt += r * qtk[(long)b * D_DIM + d];
  }
  rA[tid] = ssq; rB[tid] = dq; rC[tid] = dqt; __syncthreads();
  for (int s = 128; s > 0; s >>= 1) {
    if (tid < s) { rA[tid] += rA[tid + s]; rB[tid] += rB[tid + s]; rC[tid] += rC[tid + s]; }
    __syncthreads();
  }
  if (tid == 0) {
    float nrm = fmaxf(sqrtf(rA[0]), EPS_W);
    finalS[row] = 0.5f * (rB[0] + rC[0]) / nrm;
  }
}

// ---------------- 8. final top-3 per query; out = [ids(128x3) | scores(128x3)] ----------------
__global__ void final_top3_kernel(const float* __restrict__ finalS, const int* __restrict__ topmI,
                                  float* __restrict__ out) {
  int b = threadIdx.x;
  if (b >= B_Q) return;
  float bs[TOP_X]; int bm[TOP_X];
#pragma unroll
  for (int x = 0; x < TOP_X; ++x) { bs[x] = -3.0e38f; bm[x] = 0; }
#pragma unroll
  for (int m = 0; m < M_TOP; ++m) {
    float v = finalS[b * M_TOP + m]; int id = m;
#pragma unroll
    for (int x = 0; x < TOP_X; ++x) {
      if (v > bs[x]) { float tv = bs[x]; int ti = bm[x];
                       bs[x] = v; bm[x] = id; v = tv; id = ti; }
    }
  }
#pragma unroll
  for (int x = 0; x < TOP_X; ++x) {
    out[b * TOP_X + x]               = (float)topmI[b * M_TOP + bm[x]];
    out[B_Q * TOP_X + b * TOP_X + x] = bs[x];
  }
}

// ---------------- host orchestration ----------------
extern "C" void kernel_launch(void* const* d_in, const int* in_sizes, int n_in,
                              void* d_out, int out_size, void* d_ws, size_t ws_size,
                              hipStream_t stream) {
  (void)in_sizes; (void)n_in; (void)out_size; (void)ws_size;
  const float* q  = (const float*)d_in[0];
  const float* db = (const float*)d_in[1];
  float* out = (float*)d_out;

  char* ws = (char*)d_ws;
  size_t off = 0;
  auto take = [&](size_t bytes) -> char* {
    off = (off + 255) & ~(size_t)255;
    char* p = ws + off; off += bytes; return p;
  };
  unsigned short* dbbf  = (unsigned short*)take((size_t)N_DB * D_DIM * 2);       // 102.4 MB
  unsigned short* qbf   = (unsigned short*)take((size_t)B_Q * D_DIM * 2);
  unsigned short* a2bf  = (unsigned short*)take((size_t)B_Q * M_TOP * D_DIM * 2);
  float* Sbuf   = (float*)take((size_t)B_Q * M_TOP * CHUNK_N * 4);               // 128 MB (reused)
  float* candS  = (float*)take((size_t)B_Q * M_TOP * NCHUNKS * TKK * 4);
  int*   candI  = (int*)  take((size_t)B_Q * M_TOP * NCHUNKS * TKK * 4);
  float* topmS  = (float*)take((size_t)B_Q * TKK * 4);
  int*   topmI  = (int*)  take((size_t)B_Q * TKK * 4);
  float* s2     = (float*)take((size_t)B_Q * M_TOP * TKK * 4);
  int*   i2     = (int*)  take((size_t)B_Q * M_TOP * TKK * 4);
  float* qtk    = (float*)take((size_t)B_Q * D_DIM * 4);
  float* finalS = (float*)take((size_t)B_Q * M_TOP * 4);

  dim3 blk(256);

  // 1. bf16 conversion
  {
    long n = (long)N_DB * D_DIM;
    f32_to_bf16_kernel<<<(unsigned)((n / 4 + 255) / 256), blk, 0, stream>>>(db, dbbf, n);
    long nq = (long)B_Q * D_DIM;
    f32_to_bf16_kernel<<<(unsigned)((nq / 4 + 255) / 256), blk, 0, stream>>>(q, qbf, nq);
  }

  // 2. GEMM1 (128 x 100000) in N-chunks + per-chunk top-10
  for (int c = 0; c < NCHUNKS; ++c) {
    dim3 g((CHUNK_N + 255) / 256, B_Q / 32);
    wmma_gemm_nt_bf16<<<g, blk, 0, stream>>>(qbf, B_Q,
        dbbf + (size_t)c * CHUNK_N * D_DIM, CHUNK_N, Sbuf);
    chunk_topk_kernel<<<B_Q, blk, 0, stream>>>(Sbuf, CHUNK_N, c * CHUNK_N, c, NCHUNKS,
                                               candS, candI);
  }
  merge_cand_kernel<<<1, 256, 0, stream>>>(candS, candI, B_Q, NCHUNKS, topmS, topmI);

  // 3. epilogue prep
  build_qtk_kernel<<<B_Q, blk, 0, stream>>>(q, db, topmI, qtk);
  build_a2_kernel<<<B_Q * M_TOP, blk, 0, stream>>>(db, topmI, a2bf);

  // 4. GEMM2 (1280 x 100000) in N-chunks + per-chunk top-10
  for (int c = 0; c < NCHUNKS; ++c) {
    dim3 g((CHUNK_N + 255) / 256, (B_Q * M_TOP) / 32);
    wmma_gemm_nt_bf16<<<g, blk, 0, stream>>>(a2bf, B_Q * M_TOP,
        dbbf + (size_t)c * CHUNK_N * D_DIM, CHUNK_N, Sbuf);
    chunk_topk_kernel<<<B_Q * M_TOP, blk, 0, stream>>>(Sbuf, CHUNK_N, c * CHUNK_N, c, NCHUNKS,
                                                       candS, candI);
  }
  merge_cand_kernel<<<(B_Q * M_TOP + 255) / 256, 256, 0, stream>>>(candS, candI,
                                                                   B_Q * M_TOP, NCHUNKS, s2, i2);

  // 5. refinement + final scores + top-3
  refine_score_kernel<<<B_Q * M_TOP, blk, 0, stream>>>(q, qtk, db, s2, i2, finalS);
  final_top3_kernel<<<1, 128, 0, stream>>>(finalS, topmI, out);

  (void)topmS;
}
